// MultiHeadAttention_14637248545279
// MI455X (gfx1250) — matrix operand order repaired and checked by reading, not compiled, not run on hardware
//
#include <hip/hip_runtime.h>

// ---------------------------------------------------------------------------
// MI455X (gfx1250) fused entmax-1.5 multi-head attention
//   B=8, S=1024, E=512, H=8, DK=64
// ---------------------------------------------------------------------------

typedef __attribute__((ext_vector_type(16))) __bf16         v16bf;
typedef __attribute__((ext_vector_type(16))) unsigned short u16x16;
typedef __attribute__((ext_vector_type(8)))  float          v8f;
typedef __attribute__((ext_vector_type(4)))  unsigned int   u32x4;
typedef __attribute__((ext_vector_type(8)))  int            i32x8;
typedef __attribute__((ext_vector_type(4)))  int            i32x4;

static constexpr int Bn = 8;
static constexpr int Sn = 1024;
static constexpr int En = 512;
static constexpr int Hn = 8;
static constexpr int DKn = 64;
static constexpr float SCALEn = 0.125f;          // 1/sqrt(64)

// round-to-nearest-even f32 -> bf16 bits
static __device__ __forceinline__ unsigned short f2bf(float x) {
  unsigned int u = __builtin_bit_cast(unsigned int, x);
  unsigned int r = u + 0x7FFFu + ((u >> 16) & 1u);
  return (unsigned short)(r >> 16);
}

static __device__ __forceinline__ v8f wmma_bf16(u16x16 a, u16x16 b, v8f c) {
  return __builtin_amdgcn_wmma_f32_16x16x32_bf16(
      /*neg_a=*/false, __builtin_bit_cast(v16bf, a),
      /*neg_b=*/false, __builtin_bit_cast(v16bf, b),
      /*c_mod=*/(short)0, c, /*reuse_a=*/false, /*reuse_b=*/false);
}

// ---------------------------------------------------------------------------
// Kernel 1: projections.  Y = X @ W^T  (X:[8192,512] f32, W:[512,512] f32)
// Block = 8 waves computing a 16x128 output tile; A tile (16x512) staged once
// in LDS as bf16 (row pad 528 kills the 512-stride bank conflict) and shared
// by all 8 waves.  K-loop = 16 x (16x16x32 bf16 WMMA, f32 accum).
// Q,K stored bf16 as [b,h,s,d]; V stored bf16 TRANSPOSED as [b,h,d,s].
// ---------------------------------------------------------------------------
static constexpr int XROW = 528;                   // 512 + 16 pad (ushorts)

__global__ __launch_bounds__(256) void proj_wmma_kernel(
    const float* __restrict__ Xq, const float* __restrict__ Xk,
    const float* __restrict__ Xv, const float* __restrict__ Wq,
    const float* __restrict__ Wk, const float* __restrict__ Wv,
    unsigned short* __restrict__ Qb, unsigned short* __restrict__ Kb,
    unsigned short* __restrict__ Vt) {
  __shared__ unsigned short xs[16 * XROW];         // 16.5 KB

  const int proj = blockIdx.y;                     // 0=Q 1=K 2=V
  const float* X = (proj == 0) ? Xq : (proj == 1) ? Xk : Xv;
  const float* W = (proj == 0) ? Wq : (proj == 1) ? Wk : Wv;
  unsigned short* Out = (proj == 0) ? Qb : (proj == 1) ? Kb : Vt;
  const bool isV = (proj == 2);

  const int mt = blockIdx.x & 511;                 // 512 row tiles
  const int nb = blockIdx.x >> 9;                  // 4 col blocks of 128
  const int m0 = mt * 16;

  // ---- stage X tile [16 x 512] -> bf16 LDS (coalesced) ----
  {
    const int r  = threadIdx.x >> 4;
    const int c0 = (threadIdx.x & 15) * 32;
    const float* xp = X + (size_t)(m0 + r) * En + c0;
#pragma unroll
    for (int i = 0; i < 32; ++i) xs[r * XROW + c0 + i] = f2bf(xp[i]);
  }
  __syncthreads();

  const int lane = threadIdx.x & 31;
  const int wid  = threadIdx.x >> 5;
  const int hi   = lane >> 4;                      // half-wave select
  const int n0   = nb * 128 + wid * 16;            // this wave's 16 columns
  const int bcol = n0 + (lane & 15);               // B: column per lane
  const int am   = lane & 15;                      // A: row per lane
  const int arun = hi << 4;                        // A: K offset 0/16
  const int brun = hi << 4;                        // B: K offset 0/16

  v8f acc = {0.f, 0.f, 0.f, 0.f, 0.f, 0.f, 0.f, 0.f};
  for (int kb = 0; kb < En; kb += 32) {
    u16x16 a, bfrag;
    // A(m,k): k = kb+arun+e
    const unsigned short* ap = &xs[am * XROW + kb + arun];
#pragma unroll
    for (int e = 0; e < 8; ++e) {
      a[e]     = ap[e];
      a[e + 8] = ap[e + 8];
    }
    // B(k,n) = W[n][k]; lane holds 16 consecutive k at kb+brun
    const float* bp = W + (size_t)bcol * En + kb + brun;
#pragma unroll
    for (int e = 0; e < 16; ++e) bfrag[e] = f2bf(bp[e]);
    acc = wmma_bf16(a, bfrag, acc);
  }

  // C/D layout: n = lane&15, m = r + 8*(lane>=16).  Branch-free scatter.
#pragma unroll
  for (int r = 0; r < 8; ++r) {
    const int m  = r + (hi << 3);
    const int g  = m0 + m;                         // flattened (b,s) row
    const int bb = g >> 10, ss = g & 1023;
    const int hh = bcol >> 6, dd = bcol & 63;
    const size_t idx =
        isV ? ((size_t)(bb * Hn + hh) * DKn + dd) * Sn + ss
            : ((size_t)(bb * Hn + hh) * Sn + ss) * DKn + dd;
    Out[idx] = f2bf(acc[r]);
  }
}

// ---------------------------------------------------------------------------
// Kernel 2: fused QK^T -> entmax1.5 (bisection for tau) -> P.V
// Block = 256 threads (8 waves), one (b,h) x 16-query tile.
// Scores tile 16x1024 f32 lives in LDS.  The per-head V tile (64x1024 bf16,
// 128 KB) is DMA'd into LDS by the Tensor Data Mover, issued before phase 1
// and waited on (s_wait_tensorcnt) after the entmax phases -- the DMA runs
// entirely under the memory-free bisection compute.
// ---------------------------------------------------------------------------
static constexpr int SROW = 1028;                  // 1024 + pad (bank spread)
static constexpr int CORE_FLOATS =
    16 * SROW + 256 + 16 + 16 + 16 + 16 * 64;      // 17776 floats = 71104 B
static constexpr size_t VLDS_BYTES = (size_t)DKn * Sn * 2;     // 131072 B
static constexpr size_t SMEM_BYTES = CORE_FLOATS * sizeof(float) + VLDS_BYTES;

__global__ __launch_bounds__(256) void attn_entmax_kernel(
    const unsigned short* __restrict__ Qb, const unsigned short* __restrict__ Kb,
    const unsigned short* __restrict__ Vt, float* __restrict__ out) {
  extern __shared__ float smem[];
  float* scores = smem;                            // [16][SROW]
  float* redbuf = scores + 16 * SROW;              // [256]
  float* rowmax = redbuf + 256;                    // [16]
  float* taulo  = rowmax + 16;                     // [16]
  float* tauhi  = taulo + 16;                      // [16]
  float* outbuf = tauhi + 16;                      // [16][64]
  // TDM destination: layout [chunk c=s/16][d][16 elems] (32B cells)
  unsigned short* Vlds = (unsigned short*)(smem + CORE_FLOATS);

  const int tid  = threadIdx.x;
  const int lane = tid & 31;
  const int wid  = tid >> 5;
  const int hi   = lane >> 4;
  const int q0   = blockIdx.x * 16;
  const int bh   = blockIdx.y;                     // b*H + h

  for (int i = tid; i < 16 * 64; i += 256) outbuf[i] = 0.f;

  // ---- issue TDM: V head tile [64 d][1024 s] bf16 -> LDS ---------------
  // 3-D walk: X = 16 elems (dim0), Y = d (dim1, stride 1024),
  //           Z = 16-elem chunk (dim2, stride 16)  =>  LDS[c][d][x]
  if (wid == 0) {
    const unsigned long long ga =
        (unsigned long long)(const void*)(Vt + (size_t)bh * DKn * Sn);
    const unsigned int ldsoff =
        (unsigned int)(unsigned long long)(void*)Vlds;   // WG-relative LDS byte addr
    u32x4 g0;
    g0[0] = 1u;                                    // count=1, user descriptor
    g0[1] = ldsoff;                                // lds_addr (bytes)
    g0[2] = (unsigned int)(ga & 0xFFFFFFFFu);      // global_addr[31:0]
    g0[3] = (unsigned int)((ga >> 32) & 0x01FFFFFFu) | (2u << 30); // [56:32]|type=2
    i32x8 g1;
    g1[0] = (int)(1u << 16);                       // workgroup_mask=0, data_size=2B
    g1[1] = (int)(16u << 16);                      // tensor_dim0 = 16
    g1[2] = (int)(64u << 16);                      // tensor_dim1 = 64
    g1[3] = (int)(16u << 16);                      // tile_dim0 = 16
    g1[4] = (int)(64u | (32u << 16));              // tile_dim1=64, tile_dim2=32
    g1[5] = 1024;                                  // tensor_dim0_stride = 1024
    g1[6] = (int)(16u << 16);                      // tensor_dim1_stride = 16
    g1[7] = 0;
    i32x4 g2;
    g2[0] = 32;                                    // tensor_dim2 = 32
    g2[1] = 0;                                     // tensor_dim3 / unused
    g2[2] = 0;                                     // tensor_dim2_stride lo
    g2[3] = 0;                                     // stride hi | tile_dim3=0
    i32x4 g3 = {0, 0, 0, 0};
    i32x8 gx = {0, 0, 0, 0, 0, 0, 0, 0};
    __builtin_amdgcn_tensor_load_to_lds(g0, g1, g2, g3, gx, 0);
  }

  // ---- phase 1: scores[m][key] = (q . k) * SCALE * 0.5 (z/2 folded) ----
  const unsigned short* Qrow = Qb + (size_t)bh * Sn * DKn;
  const unsigned short* Krow = Kb + (size_t)bh * Sn * DKn;

  u16x16 afrag[2];                                 // Q tile, two K-steps of 32
  {
    const int am = lane & 15;
#pragma unroll
    for (int t = 0; t < 2; ++t) {
      const unsigned short* ap =
          Qrow + (size_t)(q0 + am) * DKn + t * 32 + (hi << 4);
#pragma unroll
      for (int e = 0; e < 8; ++e) {
        afrag[t][e]     = ap[e];
        afrag[t][e + 8] = ap[e + 8];
      }
    }
  }
  // each wave produces keys [wid*128, wid*128+128)
  for (int ntile = wid * 8; ntile < wid * 8 + 8; ++ntile) {
    const int key0 = ntile * 16;
    // prefetch next 16-key slab of K into cache (global_prefetch_b8)
    __builtin_prefetch(Krow + (size_t)(key0 + 16 + (lane & 15)) * DKn, 0, 1);
    v8f acc = {0.f, 0.f, 0.f, 0.f, 0.f, 0.f, 0.f, 0.f};
#pragma unroll
    for (int t = 0; t < 2; ++t) {
      // B(k,n)=Kproj[key=n][d=k]; lane reads 16 contiguous bf16 (32B)
      const unsigned short* bp =
          Krow + (size_t)(key0 + (lane & 15)) * DKn + t * 32 + (hi << 4);
      u16x16 bfrag = *(const u16x16*)bp;
      acc = wmma_bf16(afrag[t], bfrag, acc);
    }
#pragma unroll
    for (int r = 0; r < 8; ++r) {
      const int m = r + (hi << 3);
      scores[m * SROW + key0 + (lane & 15)] = acc[r] * (SCALEn * 0.5f);
    }
  }
  __syncthreads();

  // ---- phase 2: row maxima ---------------------------------------------
  const int row = tid >> 4;                        // 0..15
  const int seg = tid & 15;                        // 64 elems each
  {
    float m = -3.0e38f;
    const int base = row * SROW + seg * 64;
    for (int i = 0; i < 64; ++i) m = fmaxf(m, scores[base + i]);
    redbuf[tid] = m;
  }
  __syncthreads();
  if (tid < 16) {
    float m = -3.0e38f;
    for (int j = 0; j < 16; ++j) m = fmaxf(m, redbuf[tid * 16 + j]);
    rowmax[tid] = m;
    taulo[tid]  = -1.0f;                           // tau in [zmax-1, zmax-1/sqrt(K)]
    tauhi[tid]  = -0.03125f;                       // with zmax shifted to 0
  }
  __syncthreads();

  // ---- phase 3: bisection on f(tau) = sum(max(z - tau,0)^2) = 1 --------
  for (int it = 0; it < 30; ++it) {
    const float tau = 0.5f * (taulo[row] + tauhi[row]);
    const float rm  = rowmax[row];
    float s = 0.f;
    const int base = row * SROW + seg * 64;
    for (int i = 0; i < 64; ++i) {
      float d = scores[base + i] - rm - tau;
      d = fmaxf(d, 0.f);
      s += d * d;
    }
    redbuf[tid] = s;
    __syncthreads();
    if (tid < 16) {
      float f = 0.f;
      for (int j = 0; j < 16; ++j) f += redbuf[tid * 16 + j];
      const float mid = 0.5f * (taulo[tid] + tauhi[tid]);
      if (f >= 1.0f) taulo[tid] = mid; else tauhi[tid] = mid;
    }
    __syncthreads();
  }

  // ---- phase 4: probabilities overwrite scores -------------------------
  {
    const float tau = 0.5f * (taulo[row] + tauhi[row]);
    const float rm  = rowmax[row];
    const int base = row * SROW + seg * 64;
    for (int i = 0; i < 64; ++i) {
      float d = scores[base + i] - rm - tau;
      d = fmaxf(d, 0.f);
      scores[base + i] = d * d;
    }
  }
  __syncthreads();

  // V tile DMA must be complete before phase 5 consumes it
  if (wid == 0) __builtin_amdgcn_s_wait_tensorcnt(0);
  __syncthreads();

  // ---- phase 5: out(16x64) = P(16x1024) . V(1024x64) -------------------
  // wave w covers keys [w*128, w*128+128); partials summed via ds_add_f32
  const int k0w = wid * 128;
#pragma unroll
  for (int nt = 0; nt < 4; ++nt) {
    v8f acc = {0.f, 0.f, 0.f, 0.f, 0.f, 0.f, 0.f, 0.f};
    const int d = nt * 16 + (lane & 15);
#pragma unroll
    for (int j = 0; j < 4; ++j) {
      const int kb = k0w + j * 32;
      u16x16 a;
#pragma unroll
      for (int e = 0; e < 16; ++e) {
        const int kk = kb + e + (hi << 4);
        a[e] = f2bf(scores[(lane & 15) * SROW + kk]);
      }
      // B(k,n)=V[key=k][d=n]: LDS cell [c=(kb+hi*16)/16][d][0..15], 32B load
      const unsigned short* bp =
          Vlds + (size_t)((((kb >> 4) + hi) * 64 + d) << 4);
      u16x16 bfrag = *(const u16x16*)bp;
      acc = wmma_bf16(a, bfrag, acc);
    }
#pragma unroll
    for (int r = 0; r < 8; ++r) {
      const int m = r + (hi << 3);
      atomicAdd(&outbuf[m * 64 + d], acc[r]);      // ds_add_f32
    }
  }
  __syncthreads();

  // ---- writeback: out[b, q0+m, h*64 + d] -------------------------------
  const int bb = bh >> 3, hh = bh & 7;
  for (int i = tid; i < 16 * 64; i += 256) {
    const int m = i >> 6, d = i & 63;
    out[((size_t)(bb * Sn + q0 + m)) * En + hh * DKn + d] = outbuf[i];
  }
}

// ---------------------------------------------------------------------------
extern "C" void kernel_launch(void* const* d_in, const int* in_sizes, int n_in,
                              void* d_out, int out_size, void* d_ws,
                              size_t ws_size, hipStream_t stream) {
  (void)in_sizes; (void)n_in; (void)out_size; (void)ws_size;
  const float* q  = (const float*)d_in[0];
  const float* k  = (const float*)d_in[1];
  const float* v  = (const float*)d_in[2];
  const float* wq = (const float*)d_in[3];
  const float* wk = (const float*)d_in[4];
  const float* wv = (const float*)d_in[5];
  float* out = (float*)d_out;

  const size_t elems = (size_t)Bn * Hn * Sn * DKn;   // 4,194,304 per tensor
  unsigned short* Qb = (unsigned short*)d_ws;
  unsigned short* Kb = Qb + elems;
  unsigned short* Vt = Kb + elems;                   // 24 MB total bf16

  // 3 x (512 m-tiles x 4 n-blocks) blocks; 8 waves = 16x128 tile per block
  dim3 gp(2048, 3);
  proj_wmma_kernel<<<gp, 256, 0, stream>>>(q, k, v, wq, wk, wv, Qb, Kb, Vt);

  (void)hipFuncSetAttribute((const void*)attn_entmax_kernel,
                            hipFuncAttributeMaxDynamicSharedMemorySize,
                            (int)SMEM_BYTES);
  dim3 ga(Sn / 16, Bn * Hn);                         // 64 x 64 blocks
  attn_entmax_kernel<<<ga, 256, SMEM_BYTES, stream>>>(Qb, Kb, Vt, out);
}
